// MultiHeadSelfAttention_53609781788673
// MI455X (gfx1250) — matrix operand order repaired and checked by reading, not compiled, not run on hardware
//
#include <hip/hip_runtime.h>
#include <hip/hip_bf16.h>
#include <stdint.h>

#define BB   2
#define TT   2048
#define DIMc 512
#define Hc   8
#define ATTc 512
#define HDc  64

typedef __attribute__((ext_vector_type(16))) _Float16 v16h;
typedef __attribute__((ext_vector_type(8)))  float    v8f;

union Frag16 { v16h h; uint32_t u[8]; };

// ---------------- WMMA operand loaders (wave32, 16x16x32 f16) ----------------
// A-matrix 16x32 (MxK), 16-bit layout (ISA 7.12.2):
//   lane: M = lane&15 ; half = lane>>4
//   VGPR v: K = (v>=4?16:0) + (half?8:0) + 2*(v&3) + {0,1}
//   (pairs contiguous -> v0..3 / v4..7 each become one global_load_b128)
__device__ __forceinline__ v16h load_a16(const _Float16* base, int ld, int col0) {
  const int lane = threadIdx.x & 31;
  const int m  = lane & 15;
  const int hf = (lane >> 4) & 1;
  Frag16 f;
#pragma unroll
  for (int v = 0; v < 8; ++v) {
    const int k = ((v & 4) ? 16 : 0) + (hf ? 8 : 0) + 2 * (v & 3);
    f.u[v] = *(const uint32_t*)(base + m * ld + col0 + k);
  }
  return f.h;
}

// B-matrix 32x16 (KxN), 16-bit layout:
//   lane: N = lane&15 ; half = lane>>4
//   VGPR v: K = (v>=4?16:0) + (half?8:0) + 2*(v&3) + {0,1}
// Source stored [N][K] row-major (transposed weights / K / Vt).
__device__ __forceinline__ v16h load_b16(const _Float16* base, int ld, int col0) {
  const int lane = threadIdx.x & 31;
  const int n  = lane & 15;
  const int hf = (lane >> 4) & 1;
  Frag16 f;
#pragma unroll
  for (int v = 0; v < 8; ++v) {
    const int k = ((v & 4) ? 16 : 0) + (hf ? 8 : 0) + 2 * (v & 3);
    f.u[v] = *(const uint32_t*)(base + n * ld + col0 + k);
  }
  return f.h;
}

__device__ __forceinline__ v8f wmma16(v16h a, v16h b, v8f c) {
  return __builtin_amdgcn_wmma_f32_16x16x32_f16(false, a, false, b, (short)0, c,
                                                false, false);
}

// ---------------- math helpers ----------------
__device__ __forceinline__ float fast_exp(float x) {
  x = fmaxf(x, -80.0f);                       // keep bit pattern positive
  int i = (int)(12102203.0f * x) + 1064866805;
  return __int_as_float(i);
}

__device__ __forceinline__ float rowmax16(float v) {
#pragma unroll
  for (int off = 1; off < 16; off <<= 1) v = fmaxf(v, __shfl_xor(v, off, 32));
  return v;
}
__device__ __forceinline__ float rowsum16(float v) {
#pragma unroll
  for (int off = 1; off < 16; off <<= 1) v += __shfl_xor(v, off, 32);
  return v;
}

// -------- software-pipelined 16x64 GEMM core (A row-major, W [N][K]) --------
// Ping-pong double buffering (no inter-buffer register copies). Kdim must be
// a multiple of 64 (here always 512 -> 16 k-steps of 32).
__device__ __forceinline__ void gemm_16x64(const _Float16* Abase, int lda,
                                           const _Float16* Wbase, int ldb,
                                           int Kdim, v8f acc[4]) {
  v16h a0, b0[4], a1, b1[4];
  a0 = load_a16(Abase, lda, 0);
#pragma unroll
  for (int t = 0; t < 4; ++t)
    b0[t] = load_b16(Wbase + (size_t)(t * 16) * ldb, ldb, 0);

  int k0 = 0;
  for (; k0 + 64 < Kdim; k0 += 64) {
    // prefetch step k0+32 into buffer 1, compute with buffer 0
    a1 = load_a16(Abase, lda, k0 + 32);
#pragma unroll
    for (int t = 0; t < 4; ++t)
      b1[t] = load_b16(Wbase + (size_t)(t * 16) * ldb, ldb, k0 + 32);
#pragma unroll
    for (int t = 0; t < 4; ++t) acc[t] = wmma16(a0, b0[t], acc[t]);

    // prefetch step k0+64 into buffer 0, compute with buffer 1
    a0 = load_a16(Abase, lda, k0 + 64);
#pragma unroll
    for (int t = 0; t < 4; ++t)
      b0[t] = load_b16(Wbase + (size_t)(t * 16) * ldb, ldb, k0 + 64);
#pragma unroll
    for (int t = 0; t < 4; ++t) acc[t] = wmma16(a1, b1[t], acc[t]);
  }
  // tail: steps k0 and k0+32
  a1 = load_a16(Abase, lda, k0 + 32);
#pragma unroll
  for (int t = 0; t < 4; ++t)
    b1[t] = load_b16(Wbase + (size_t)(t * 16) * ldb, ldb, k0 + 32);
#pragma unroll
  for (int t = 0; t < 4; ++t) acc[t] = wmma16(a0, b0[t], acc[t]);
#pragma unroll
  for (int t = 0; t < 4; ++t) acc[t] = wmma16(a1, b1[t], acc[t]);
}

// ---------------- kernel 0a: f32 -> f16 ----------------
__global__ void k_cvt_f16(const float* __restrict__ src,
                          _Float16* __restrict__ dst, int n) {
  int i = blockIdx.x * blockDim.x + threadIdx.x;
  if (i < n) dst[i] = (_Float16)src[i];
}

// ---------------- kernel 0b: W[K][N] f32 -> Wt[N][K] f16 ----------------
__global__ void k_transpose_w(const float* __restrict__ W,
                              _Float16* __restrict__ Wt, int rows, int cols) {
  int i = blockIdx.x * blockDim.x + threadIdx.x;
  if (i < rows * cols) {
    int r = i / cols, c = i % cols;
    Wt[c * rows + r] = (_Float16)W[r * cols + c];
  }
}

// ---------------- kernel 1: fused QKV projection ----------------
__global__ __launch_bounds__(256) void k_qkv(
    const _Float16* __restrict__ x16, const _Float16* __restrict__ Wqt,
    const _Float16* __restrict__ Wkt, const _Float16* __restrict__ Wvt,
    const float* __restrict__ bq, const float* __restrict__ bk,
    const float* __restrict__ bv, _Float16* __restrict__ Qd,
    _Float16* __restrict__ Kd, _Float16* __restrict__ Vtd) {
  const int wg    = blockIdx.x * (blockDim.x >> 5) + (threadIdx.x >> 5);
  const int mat   = wg >> 11;        // 0=Q 1=K 2=V
  const int rem   = wg & 2047;
  const int mtile = rem & 255;       // M = B*T = 4096 -> 256 tiles of 16
  const int n64   = rem >> 8;        // 8 groups of 64 output columns
  const _Float16* Wt  = (mat == 0) ? Wqt : (mat == 1) ? Wkt : Wvt;
  const float*    bia = (mat == 0) ? bq  : (mat == 1) ? bk  : bv;

  const int lane = threadIdx.x & 31;
  const int n = lane & 15, hf = (lane >> 4) & 1;

  v8f acc[4] = {};
  gemm_16x64(x16 + (size_t)mtile * 16 * DIMc, DIMc,
             Wt + (size_t)(n64 * 64) * DIMc, DIMc, DIMc, acc);

#pragma unroll
  for (int t = 0; t < 4; ++t) {
    const int   ncol = n64 * 64 + t * 16 + n;
    const float bval = bia[ncol];
    const int   hh = ncol >> 6, dd = ncol & 63;
#pragma unroll
    for (int r = 0; r < 8; ++r) {
      const int mrow = mtile * 16 + r + 8 * hf;
      const int bidx = mrow >> 11;       // / T
      const int trow = mrow & 2047;      // % T
      const _Float16 hv = (_Float16)(acc[t][r] + bval);
      if (mat == 2)
        Vtd[(size_t)((bidx * Hc + hh) * HDc + dd) * TT + trow] = hv;
      else if (mat == 1)
        Kd[(size_t)((bidx * Hc + hh) * TT + trow) * HDc + dd] = hv;
      else
        Qd[(size_t)((bidx * Hc + hh) * TT + trow) * HDc + dd] = hv;
    }
  }
}

// ---------------- kernel 2: flash attention (one wave per 16-query tile) ----
struct KFrags { v16h f0, f1, f2, f3; };

__device__ __forceinline__ void load_kfrags(KFrags& kf, const _Float16* Kb,
                                            int kblk) {
  const _Float16* K0 = Kb + (size_t)(kblk * 32) * HDc;
  const _Float16* K1 = Kb + (size_t)(kblk * 32 + 16) * HDc;
  kf.f0 = load_b16(K0, HDc, 0);
  kf.f1 = load_b16(K0, HDc, 32);
  kf.f2 = load_b16(K1, HDc, 0);
  kf.f3 = load_b16(K1, HDc, 32);
}

// One 32-key block: scores + online softmax + P@V, prefetching next K tile
// into `nxt` while this block's VALU work runs.
__device__ __forceinline__ void attn_block(
    int kb, int kn, const _Float16* Kb, const _Float16* Vb, v16h qa0, v16h qa1,
    KFrags& cur, KFrags& nxt, float scaling, float* mrow, float* lrow,
    v8f* acc, _Float16* pslice, int n, int hf) {
  // V fragments for this block: independent of scores, issue early
  v16h vb[4];
#pragma unroll
  for (int t = 0; t < 4; ++t)
    vb[t] = load_b16(Vb + (size_t)(t * 16) * TT, TT, kb * 32);

  v8f s0 = {}, s1 = {};
  s0 = wmma16(qa0, cur.f0, s0);
  s0 = wmma16(qa1, cur.f1, s0);
  s1 = wmma16(qa0, cur.f2, s1);
  s1 = wmma16(qa1, cur.f3, s1);

  // prefetch next block's K fragments into the other buffer (ping-pong)
  load_kfrags(nxt, Kb, kn);

  // online softmax (VALU + 16-lane shuffles overlap in-flight loads)
  float p0[8], p1[8], alpha[8];
#pragma unroll
  for (int r = 0; r < 8; ++r) {
    const float a  = s0[r] * scaling;
    const float b  = s1[r] * scaling;
    const float tm = rowmax16(fmaxf(a, b));
    const float mn = fmaxf(mrow[r], tm);
    alpha[r] = fast_exp(mrow[r] - mn);
    p0[r]    = fast_exp(a - mn);
    p1[r]    = fast_exp(b - mn);
    lrow[r]  = lrow[r] * alpha[r] + rowsum16(p0[r] + p1[r]);
    mrow[r]  = mn;
  }
#pragma unroll
  for (int t = 0; t < 4; ++t)
#pragma unroll
    for (int r = 0; r < 8; ++r) acc[t][r] *= alpha[r];

  // C-layout f32 P -> row-major 16x32 f16 in LDS (per-wave slice)
#pragma unroll
  for (int r = 0; r < 8; ++r) {
    const int mr = r + 8 * hf;
    pslice[mr * 32 + n]      = (_Float16)p0[r];
    pslice[mr * 32 + 16 + n] = (_Float16)p1[r];
  }
#if __has_builtin(__builtin_amdgcn_s_wait_dscnt)
  __builtin_amdgcn_s_wait_dscnt(0);
#else
  asm volatile("s_wait_dscnt 0x0" ::: "memory");
#endif
  const v16h pa = load_a16(pslice, 32, 0);   // P[16, k 0..31]
#pragma unroll
  for (int t = 0; t < 4; ++t) acc[t] = wmma16(pa, vb[t], acc[t]);
}

__global__ __launch_bounds__(256) void k_attn(
    const _Float16* __restrict__ Qd, const _Float16* __restrict__ Kd,
    const _Float16* __restrict__ Vtd, _Float16* __restrict__ O16,
    float scaling) {
  __shared__ _Float16 pbuf[8][16 * 32];   // per-wave P tile staging (8 KB)
  const int w  = threadIdx.x >> 5;
  const int wg = blockIdx.x * (blockDim.x >> 5) + w;
  const int qt = wg & 127;
  const int hh = (wg >> 7) & 7;
  const int bb = wg >> 10;
  const int lane = threadIdx.x & 31;
  const int n = lane & 15, hf = (lane >> 4) & 1;

  const _Float16* Qb = Qd + (size_t)((bb * Hc + hh) * TT + qt * 16) * HDc;
  const _Float16* Kb = Kd + (size_t)(bb * Hc + hh) * TT * HDc;
  const _Float16* Vb = Vtd + (size_t)(bb * Hc + hh) * HDc * TT;

  const v16h qa0 = load_a16(Qb, HDc, 0);    // Q[16, d 0..31]
  const v16h qa1 = load_a16(Qb, HDc, 32);   // Q[16, d 32..63]

  float mrow[8], lrow[8];
#pragma unroll
  for (int r = 0; r < 8; ++r) { mrow[r] = -50.0f; lrow[r] = 0.0f; }
  v8f acc[4] = {};

  const int NKB = TT / 32;   // 64, even
  KFrags A, B;
  load_kfrags(A, Kb, 0);

  for (int kb = 0; kb < NKB; kb += 2) {
    attn_block(kb, kb + 1, Kb, Vb, qa0, qa1, A, B, scaling, mrow, lrow, acc,
               &pbuf[w][0], n, hf);
    const int kn2 = (kb + 2 < NKB) ? kb + 2 : kb + 1;   // clamped, branch-free
    attn_block(kb + 1, kn2, Kb, Vb, qa0, qa1, B, A, scaling, mrow, lrow, acc,
               &pbuf[w][0], n, hf);
  }

#pragma unroll
  for (int r = 0; r < 8; ++r) {
    const float inv  = 1.0f / lrow[r];
    const int   trow = qt * 16 + r + 8 * hf;
#pragma unroll
    for (int t = 0; t < 4; ++t) {
      O16[(size_t)(bb * TT + trow) * ATTc + hh * HDc + t * 16 + n] =
          (_Float16)(acc[t][r] * inv);
    }
  }
}

// ---------------- kernel 3: output projection O16 @ Wo -> f32 out ----------
__global__ __launch_bounds__(256) void k_oproj(
    const _Float16* __restrict__ O16, const _Float16* __restrict__ Wot,
    float* __restrict__ out) {
  const int wg    = blockIdx.x * (blockDim.x >> 5) + (threadIdx.x >> 5);
  const int mtile = wg & 255;
  const int n64   = wg >> 8;
  const int lane  = threadIdx.x & 31;
  const int n = lane & 15, hf = (lane >> 4) & 1;

  v8f acc[4] = {};
  gemm_16x64(O16 + (size_t)mtile * 16 * ATTc, ATTc,
             Wot + (size_t)(n64 * 64) * ATTc, ATTc, ATTc, acc);

#pragma unroll
  for (int t = 0; t < 4; ++t) {
    const int ncol = n64 * 64 + t * 16 + n;
#pragma unroll
    for (int r = 0; r < 8; ++r) {
      const int mrow = mtile * 16 + r + 8 * hf;
      out[(size_t)mrow * DIMc + ncol] = acc[t][r];
    }
  }
}

// ---------------- host side ----------------
static inline float q_rsqrt_host(float x) {
  union { float f; int32_t i; } u;
  u.f = x;
  u.i = 0x5F3759DF - (u.i >> 1);
  float y = u.f;
  return y * (1.5f - 0.5f * x * y * y);
}

extern "C" void kernel_launch(void* const* d_in, const int* in_sizes, int n_in,
                              void* d_out, int out_size, void* d_ws,
                              size_t ws_size, hipStream_t stream) {
  const float* x  = (const float*)d_in[0];
  const float* Wq = (const float*)d_in[1];
  const float* bq = (const float*)d_in[2];
  const float* Wk = (const float*)d_in[3];
  const float* bk = (const float*)d_in[4];
  const float* Wv = (const float*)d_in[5];
  const float* bv = (const float*)d_in[6];
  const float* Wo = (const float*)d_in[7];
  float* out = (float*)d_out;

  char* p = (char*)d_ws;
  _Float16* x16 = (_Float16*)p; p += (size_t)BB * TT * DIMc * 2;
  _Float16* Wqt = (_Float16*)p; p += (size_t)DIMc * ATTc * 2;
  _Float16* Wkt = (_Float16*)p; p += (size_t)DIMc * ATTc * 2;
  _Float16* Wvt = (_Float16*)p; p += (size_t)DIMc * ATTc * 2;
  _Float16* Wot = (_Float16*)p; p += (size_t)ATTc * DIMc * 2;
  _Float16* Qd  = (_Float16*)p; p += (size_t)BB * Hc * TT * HDc * 2;
  _Float16* Kd  = (_Float16*)p; p += (size_t)BB * Hc * TT * HDc * 2;
  _Float16* Vtd = (_Float16*)p; p += (size_t)BB * Hc * HDc * TT * 2;
  _Float16* O16 = (_Float16*)p; p += (size_t)BB * TT * ATTc * 2;

  const int nX = BB * TT * DIMc;
  k_cvt_f16<<<(nX + 255) / 256, 256, 0, stream>>>(x, x16, nX);
  const int nW = DIMc * ATTc;
  k_transpose_w<<<(nW + 255) / 256, 256, 0, stream>>>(Wq, Wqt, DIMc, ATTc);
  k_transpose_w<<<(nW + 255) / 256, 256, 0, stream>>>(Wk, Wkt, DIMc, ATTc);
  k_transpose_w<<<(nW + 255) / 256, 256, 0, stream>>>(Wv, Wvt, DIMc, ATTc);
  k_transpose_w<<<(nW + 255) / 256, 256, 0, stream>>>(Wo, Wot, ATTc, DIMc);

  k_qkv<<<768, 256, 0, stream>>>(x16, Wqt, Wkt, Wvt, bq, bk, bv, Qd, Kd, Vtd);

  const float scaling = q_rsqrt_host(64.0f);  // Quake rsqrt(HD), as reference
  k_attn<<<256, 256, 0, stream>>>(Qd, Kd, Vtd, O16, scaling);
  k_oproj<<<256, 256, 0, stream>>>(O16, Wot, out);
}
